// ScaledDotProductAttention_76862734729512
// MI455X (gfx1250) — matrix operand order repaired and checked by reading, not compiled, not run on hardware
//
#include <hip/hip_runtime.h>
#include <hip/hip_bf16.h>

#define HEADS   8
#define QLEN    1024
#define KLEN    1024
#define DKD     64
#define QT      16          // q rows per workgroup
#define KT      32          // phase-B k tile (== WMMA bf16 K)
#define LN_EPS  1e-5f

typedef __attribute__((ext_vector_type(16))) __bf16 v16bf;
typedef __attribute__((ext_vector_type(8)))  float  v8f;

// native bf16 convert (lowers to v_cvt_pk_bf16_f32 for adjacent pairs)
__device__ __forceinline__ __bf16 f2bf(float f) { return (__bf16)f; }

__device__ __forceinline__ v8f zero8() {
    return (v8f){0.f, 0.f, 0.f, 0.f, 0.f, 0.f, 0.f, 0.f};
}

// 3-way max in one VOP3 (avoids fmaxf NaN-canonicalization self-maxes)
__device__ __forceinline__ float max3f(float a, float b, float c) {
    float r;
    asm("v_max3_num_f32 %0, %1, %2, %3" : "=v"(r) : "v"(a), "v"(b), "v"(c));
    return r;
}

// ---- fused DPP16 reductions: one VOP2+DPP instruction per butterfly stage,
//      confined to 16-lane rows (matches the WMMA C-layout half-waves) ----
__device__ __forceinline__ float hmax16(float x) {
    float t;
    asm("v_max_num_f32_dpp %0, %1, %1 quad_perm:[1,0,3,2] row_mask:0xf bank_mask:0xf bound_ctrl:1"
        : "=v"(t) : "v"(x));
    asm("v_max_num_f32_dpp %0, %1, %1 quad_perm:[2,3,0,1] row_mask:0xf bank_mask:0xf bound_ctrl:1"
        : "=v"(x) : "v"(t));
    asm("v_max_num_f32_dpp %0, %1, %1 row_half_mirror row_mask:0xf bank_mask:0xf bound_ctrl:1"
        : "=v"(t) : "v"(x));
    asm("v_max_num_f32_dpp %0, %1, %1 row_mirror row_mask:0xf bank_mask:0xf bound_ctrl:1"
        : "=v"(x) : "v"(t));
    return x;
}
__device__ __forceinline__ float hsum16(float x) {
    float t;
    asm("v_add_f32_dpp %0, %1, %1 quad_perm:[1,0,3,2] row_mask:0xf bank_mask:0xf bound_ctrl:1"
        : "=v"(t) : "v"(x));
    asm("v_add_f32_dpp %0, %1, %1 quad_perm:[2,3,0,1] row_mask:0xf bank_mask:0xf bound_ctrl:1"
        : "=v"(x) : "v"(t));
    asm("v_add_f32_dpp %0, %1, %1 row_half_mirror row_mask:0xf bank_mask:0xf bound_ctrl:1"
        : "=v"(t) : "v"(x));
    asm("v_add_f32_dpp %0, %1, %1 row_mirror row_mask:0xf bank_mask:0xf bound_ctrl:1"
        : "=v"(x) : "v"(t));
    return x;
}

// Fused DeepViT re-attention:
//   scores = QK^T/8 -> softmax(k) -> cross-head mix (8x8) -> LN(head) ->
//   att_map = head-mean ; out = attn'' @ V
// grid = (QLEN/QT, B), block = 256 (8 waves; wave w == head w)
// Softmax runs in base-2: Q is pre-scaled by log2(e)/sqrt(dk) at bf16-convert.
__global__ __launch_bounds__(256)
void deepvit_reattn_fused(const float* __restrict__ Q,
                          const float* __restrict__ K,
                          const float* __restrict__ V,
                          const float* __restrict__ Wha,  // (H,H) row-major
                          const float* __restrict__ lnw,
                          const float* __restrict__ lnb,
                          float* __restrict__ out,        // (64,1024,64)
                          float* __restrict__ amap)       // (8,1024,1024)
{
    __shared__ float  sW[HEADS * HEADS];
    __shared__ float  sLw[HEADS], sLb[HEADS];
    __shared__ float  rawA[HEADS][QT][KT];   // softmaxed attn, fp32  (16 KB)
    __shared__ __bf16 mixA[HEADS][QT][KT];   // re-attended, bf16     (8 KB)

    const int tid  = threadIdx.x;
    const int wave = tid >> 5;        // head index
    const int lane = tid & 31;
    const int l16  = lane & 15;
    const int hi   = lane >> 4;       // half-wave: 0 or 1

    const int q0 = blockIdx.x * QT;
    const int b  = blockIdx.y;
    const int bh = b * HEADS + wave;

    if (tid < HEADS * HEADS) sW[tid] = Wha[tid];
    if (tid < HEADS) { sLw[tid] = lnw[tid]; sLb[tid] = lnb[tid]; }
    __syncthreads();

    // ---- Q tile (16x64) as two bf16 A-operands, pre-scaled ----
    // A 16x32 bf16 layout: lane -> M=l16 ; element e -> K = hi*16 + e
    const float SCALE = 0.125f * 1.4426950408889634f;   // log2(e)/sqrt(dk)
    const float* Qbase = Q + ((size_t)bh * QLEN + q0) * DKD;
    v16bf qa[2];
#pragma unroll
    for (int c = 0; c < 2; ++c) {
        const float* qrow = Qbase + (size_t)l16 * DKD + c * 32 + hi * 16;
#pragma unroll
        for (int e = 0; e < 16; ++e)
            qa[c][e] = f2bf(qrow[e] * SCALE);
    }

    // ======== Phase A: online softmax stats, 64 k-columns / iteration ====
    // row of (lane,j): M = j + 8*hi ; stats replicated across each half.
    const float* Kbase = K + (size_t)bh * KLEN * DKD;
    float m[8], s[8];
#pragma unroll
    for (int j = 0; j < 8; ++j) { m[j] = -1e30f; s[j] = 0.f; }

    for (int kt = 0; kt < KLEN / 64; ++kt) {
        const int k0 = kt * 64;
        if (kt + 1 < KLEN / 64)
            __builtin_prefetch(Kbase + (size_t)(k0 + 64 + l16) * DKD, 0, 0);
        v8f sc[4];
#pragma unroll
        for (int t = 0; t < 4; ++t) sc[t] = zero8();
        // c-outer: 3 independent WMMAs between each dependent D->C pair
#pragma unroll
        for (int c = 0; c < 2; ++c) {
#pragma unroll
            for (int t = 0; t < 4; ++t) {
                // B = K^T tile: lane -> N(=k col)=l16 ; element e -> K(=d)=hi*16+e
                const float* kp = Kbase + (size_t)(k0 + t * 16 + l16) * DKD + c * 32 + hi * 16;
                v16bf kb;
#pragma unroll
                for (int e = 0; e < 16; ++e) kb[e] = f2bf(kp[e]);
                sc[t] = __builtin_amdgcn_wmma_f32_16x16x32_bf16(
                            false, qa[c], false, kb, (short)0, sc[t], false, false);
            }
        }
#pragma unroll
        for (int j = 0; j < 8; ++j) {
            // m[j] replicated per half: fold it into the butterfly input
            float w  = max3f(sc[0][j], sc[1][j], sc[2][j]);
            float mn = hmax16(max3f(w, sc[3][j], m[j]));
            float p  = __builtin_amdgcn_exp2f(sc[0][j] - mn)
                     + __builtin_amdgcn_exp2f(sc[1][j] - mn)
                     + __builtin_amdgcn_exp2f(sc[2][j] - mn)
                     + __builtin_amdgcn_exp2f(sc[3][j] - mn);
            s[j] = s[j] * __builtin_amdgcn_exp2f(m[j] - mn) + hsum16(p);
            m[j] = mn;
        }
    }
    float is[8];
#pragma unroll
    for (int j = 0; j < 8; ++j) is[j] = 1.0f / s[j];   // hoisted out of phase B

    // ================= Phase B: normalize -> mix -> LN -> @V ============
    v8f acc[4];
#pragma unroll
    for (int n = 0; n < 4; ++n) acc[n] = zero8();

    const float* Vbase = V + (size_t)bh * KLEN * DKD;
    float* amapBase = amap + ((size_t)b * QLEN + q0) * KLEN;

    for (int kt = 0; kt < KLEN / KT; ++kt) {
        const int k0 = kt * KT;

        // recompute the 16x32 score tile (c-outer to interleave the two
        // accumulator chains), normalize with (m, 1/s), stash in LDS
        v8f sc[2];
#pragma unroll
        for (int t = 0; t < 2; ++t) sc[t] = zero8();
#pragma unroll
        for (int c = 0; c < 2; ++c) {
#pragma unroll
            for (int t = 0; t < 2; ++t) {
                const float* kp = Kbase + (size_t)(k0 + t * 16 + l16) * DKD + c * 32 + hi * 16;
                v16bf kb;
#pragma unroll
                for (int e = 0; e < 16; ++e) kb[e] = f2bf(kp[e]);
                sc[t] = __builtin_amdgcn_wmma_f32_16x16x32_bf16(
                            false, qa[c], false, kb, (short)0, sc[t], false, false);
            }
        }
#pragma unroll
        for (int t = 0; t < 2; ++t)
#pragma unroll
            for (int j = 0; j < 8; ++j)
                rawA[wave][j + hi * 8][t * 16 + l16] =
                    __builtin_amdgcn_exp2f(sc[t][j] - m[j]) * is[j];
        __syncthreads();

        // cross-head mix + LN + att_map: 512 elems, 1 per thread per pass
#pragma unroll
        for (int r = 0; r < 2; ++r) {
            const int e  = r * 256 + tid;
            const int qq = e >> 5;
            const int kk = e & 31;
            float rv[8];
#pragma unroll
            for (int h = 0; h < 8; ++h) rv[h] = rawA[h][qq][kk];
            float mix[8], mu = 0.f;
#pragma unroll
            for (int h = 0; h < 8; ++h) {
                float a2 = 0.f;
#pragma unroll
                for (int hp = 0; hp < 8; ++hp) a2 = fmaf(rv[hp], sW[hp * 8 + h], a2);
                mix[h] = a2; mu += a2;
            }
            mu *= 0.125f;
            float var = 0.f;
#pragma unroll
            for (int h = 0; h < 8; ++h) { float d = mix[h] - mu; var = fmaf(d, d, var); }
            const float rs = rsqrtf(var * 0.125f + LN_EPS);
            float am = 0.f;
#pragma unroll
            for (int h = 0; h < 8; ++h) {
                float y = (mix[h] - mu) * rs * sLw[h] + sLb[h];
                mixA[h][qq][kk] = f2bf(y);
                am += y;
            }
            __builtin_nontemporal_store(am * 0.125f,
                                        amapBase + (size_t)qq * KLEN + k0 + kk);
        }
        __syncthreads();

        // A = re-attended tile for this head (16 q x 32 k) from LDS
        v16bf aa;
#pragma unroll
        for (int e = 0; e < 16; ++e)
            aa[e] = mixA[wave][l16][hi * 16 + e];

        // B = V tile (32 k x 16 d), four d-chunks; accumulate out tile
#pragma unroll
        for (int n = 0; n < 4; ++n) {
            v16bf vb;
#pragma unroll
            for (int e = 0; e < 16; ++e)
                vb[e] = f2bf(Vbase[(size_t)(k0 + hi * 16 + e) * DKD + n * 16 + l16]);
            acc[n] = __builtin_amdgcn_wmma_f32_16x16x32_bf16(
                         false, aa, false, vb, (short)0, acc[n], false, false);
        }
        __syncthreads();
    }

    // ---- epilogue: out[bh, q0+M, d] from C layout (M=j+8*hi, N=l16) ----
    float* obase = out + ((size_t)bh * QLEN + q0) * DKD;
#pragma unroll
    for (int n = 0; n < 4; ++n)
#pragma unroll
        for (int j = 0; j < 8; ++j)
            __builtin_nontemporal_store(acc[n][j],
                obase + (size_t)(j + hi * 8) * DKD + n * 16 + l16);
}

extern "C" void kernel_launch(void* const* d_in, const int* in_sizes, int n_in,
                              void* d_out, int out_size, void* d_ws, size_t ws_size,
                              hipStream_t stream) {
    const float* Q   = (const float*)d_in[0];
    const float* K   = (const float*)d_in[1];
    const float* V   = (const float*)d_in[2];
    const float* Wha = (const float*)d_in[3];
    const float* lnw = (const float*)d_in[4];
    const float* lnb = (const float*)d_in[5];

    float* out  = (float*)d_out;                         // (64,1024,64)
    float* amap = out + (size_t)64 * QLEN * DKD;         // (8,1024,1024)

    dim3 grid(QLEN / QT, 8);   // 64 q-tiles x 8 batches = 512 WGs
    deepvit_reattn_fused<<<grid, 256, 0, stream>>>(Q, K, V, Wha, lnw, lnb, out, amap);
}
